// EncoderVDP_85255100826169
// MI455X (gfx1250) — matrix operand order repaired and checked by reading, not compiled, not run on hardware
//
#include <hip/hip_runtime.h>

#define TOL   1e-3f
#define LNEPS 1e-5f

#define CB   8
#define CT   512
#define CH   8
#define CDH  32
#define CK   256
#define CE0  512
#define CDIN 256
#define CM   (CB * CT)   // 4096 rows

typedef __attribute__((ext_vector_type(16))) __bf16 v16bf;
typedef __attribute__((ext_vector_type(8)))  float  v8f;

// ---------------------------------------------------------------------------
// Unified VDP GEMM:  mu = A·B0 (+bias)(+pos)(+Rmu)
//                    var = [Avar·B1] + (A∘A)·B2 (+vbias) -> clamp/scale/relu/residual
// B matrices are bf16, layout B[n*ldb + k] (K contiguous per output column).
// Per-z base offsets: base = (z/Hdiv)*Outer + (z%Hdiv)*Inner  (elements).
// Each wave computes a 16 x (16*NT) output strip: A fragments (incl. the fp32->
// bf16 converts and squaring) are reused across NT column sub-tiles, and all
// B fragments of a k-step are loaded into distinct registers up-front so the
// backend can emit one load clause and issue the WMMA chain back-to-back.
// ---------------------------------------------------------------------------
struct GemmVdp {
  const float*  Amu;
  const float*  Avar;     // nullptr => input variance is exactly 0
  const __bf16* B0;
  const __bf16* B1;
  const __bf16* B2;
  float* Cmu;
  float* Cvar;
  const float* bias;      // nullable, [N]
  const float* vbias;     // nullable, [N]
  const float* Rmu;       // nullable residual (same base/ld as C)
  const float* Rvar;
  const float* pos;       // nullable, mu += pos[(m%Tmod)*N + n]
  int M, N, Kd, lda, ldb, ldc;
  long aO, aI, bO, bI, cO, cI;
  int   Hdiv;
  float scale;            // mu*=scale; var*=scale^2 (post-clamp)
  int   relu;
  int   Tmod;
};

__device__ inline v16bf cvt16(const float4& a0, const float4& a1,
                              const float4& a2, const float4& a3) {
  v16bf f;
  f[0]=(__bf16)a0.x;  f[1]=(__bf16)a0.y;  f[2]=(__bf16)a0.z;  f[3]=(__bf16)a0.w;
  f[4]=(__bf16)a1.x;  f[5]=(__bf16)a1.y;  f[6]=(__bf16)a1.z;  f[7]=(__bf16)a1.w;
  f[8]=(__bf16)a2.x;  f[9]=(__bf16)a2.y;  f[10]=(__bf16)a2.z; f[11]=(__bf16)a2.w;
  f[12]=(__bf16)a3.x; f[13]=(__bf16)a3.y; f[14]=(__bf16)a3.z; f[15]=(__bf16)a3.w;
  return f;
}

__device__ inline v16bf cvt16sq(const float4& a0, const float4& a1,
                                const float4& a2, const float4& a3) {
  v16bf f;
  f[0]=(__bf16)(a0.x*a0.x);  f[1]=(__bf16)(a0.y*a0.y);  f[2]=(__bf16)(a0.z*a0.z);  f[3]=(__bf16)(a0.w*a0.w);
  f[4]=(__bf16)(a1.x*a1.x);  f[5]=(__bf16)(a1.y*a1.y);  f[6]=(__bf16)(a1.z*a1.z);  f[7]=(__bf16)(a1.w*a1.w);
  f[8]=(__bf16)(a2.x*a2.x);  f[9]=(__bf16)(a2.y*a2.y);  f[10]=(__bf16)(a2.z*a2.z); f[11]=(__bf16)(a2.w*a2.w);
  f[12]=(__bf16)(a3.x*a3.x); f[13]=(__bf16)(a3.y*a3.y); f[14]=(__bf16)(a3.z*a3.z); f[15]=(__bf16)(a3.w*a3.w);
  return f;
}

template <int NT>
__global__ __launch_bounds__(256) void gemm_vdp_kernel(GemmVdp g) {
  const int lane   = threadIdx.x & 31;
  const int wave   = threadIdx.x >> 5;
  const int tilesN = g.N / (16 * NT);
  const int total  = (g.M >> 4) * tilesN;
  const int tile   = blockIdx.x * 8 + wave;
  if (tile >= total) return;                 // wave-uniform; no barriers below
  const int n0 = (tile % tilesN) * (16 * NT);
  const int m0 = (tile / tilesN) << 4;

  const int z = blockIdx.y;
  const long bA = (long)(z / g.Hdiv) * g.aO + (long)(z % g.Hdiv) * g.aI;
  const long bB = (long)(z / g.Hdiv) * g.bO + (long)(z % g.Hdiv) * g.bI;
  const long bC = (long)(z / g.Hdiv) * g.cO + (long)(z % g.Hdiv) * g.cI;

  const int half = lane >> 4;                // ISA A layout: lanes 0-15 / 16-31
  const int rq   = lane & 15;
  const int row  = m0 + rq;                  // A: M = lane%16
  const int col  = n0 + rq;                  // B/C: N = lane%16 (+ 16*nt)

  const float*  Ap  = g.Amu + bA + (long)row * g.lda;
  const float*  Vp  = g.Avar ? (g.Avar + bA + (long)row * g.lda) : nullptr;
  const __bf16* B0p = g.B0 + bB + (long)col * g.ldb;
  const __bf16* B1p = g.B1 + bB + (long)col * g.ldb;
  const __bf16* B2p = g.B2 + bB + (long)col * g.ldb;
  const long nstep = (long)16 * g.ldb;       // sub-tile column pointer step

  v8f accM[NT];
  v8f accV[NT];
  #pragma unroll
  for (int nt = 0; nt < NT; ++nt) {
    accM[nt] = (v8f){0.f,0.f,0.f,0.f,0.f,0.f,0.f,0.f};
    accV[nt] = (v8f){0.f,0.f,0.f,0.f,0.f,0.f,0.f,0.f};
  }
  const bool hasV = (g.Avar != nullptr);

  for (int k0 = 0; k0 < g.Kd; k0 += 32) {
    const int kA = k0 + half * 16;           // 16 contiguous K per lane
    const int kB = k0 + half * 16;           // 16 contiguous K per lane

    // ---- issue ALL loads of this k-step first (A fp32 + every B fragment) ----
    float4 a0 = *(const float4*)(Ap + kA);
    float4 a1 = *(const float4*)(Ap + kA + 4);
    float4 a2 = *(const float4*)(Ap + kA + 8);
    float4 a3 = *(const float4*)(Ap + kA + 12);

    v16bf fb0[NT], fb2[NT];
    #pragma unroll
    for (int nt = 0; nt < NT; ++nt) {
      fb0[nt] = *(const v16bf*)(B0p + nt * nstep + kB);
      fb2[nt] = *(const v16bf*)(B2p + nt * nstep + kB);
    }
    float4 v0, v1, v2, v3;
    v16bf fb1[NT];
    if (hasV) {
      v0 = *(const float4*)(Vp + kA);
      v1 = *(const float4*)(Vp + kA + 4);
      v2 = *(const float4*)(Vp + kA + 8);
      v3 = *(const float4*)(Vp + kA + 12);
      #pragma unroll
      for (int nt = 0; nt < NT; ++nt)
        fb1[nt] = *(const v16bf*)(B1p + nt * nstep + kB);
    }

    // ---- convert A side once, then drain the WMMA chain ----
    v16bf fm = cvt16(a0, a1, a2, a3);
    v16bf fs = cvt16sq(a0, a1, a2, a3);
    #pragma unroll
    for (int nt = 0; nt < NT; ++nt) {
      accM[nt] = __builtin_amdgcn_wmma_f32_16x16x32_bf16(false, fm, false, fb0[nt],
                                                         (short)0, accM[nt], false, false);
      accV[nt] = __builtin_amdgcn_wmma_f32_16x16x32_bf16(false, fs, false, fb2[nt],
                                                         (short)0, accV[nt], false, false);
    }
    if (hasV) {
      v16bf fv = cvt16(v0, v1, v2, v3);
      #pragma unroll
      for (int nt = 0; nt < NT; ++nt) {
        accV[nt] = __builtin_amdgcn_wmma_f32_16x16x32_bf16(false, fv, false, fb1[nt],
                                                           (short)0, accV[nt], false, false);
      }
    }
  }

  const float s2 = g.scale * g.scale;
  float* Cm = g.Cmu + bC;
  float* Cv = g.Cvar + bC;
  const float* Rm = g.Rmu  ? g.Rmu  + bC : nullptr;
  const float* Rv = g.Rvar ? g.Rvar + bC : nullptr;

  #pragma unroll
  for (int nt = 0; nt < NT; ++nt) {
    const int cn  = col + nt * 16;
    const float bmu = g.bias  ? g.bias[cn]  : 0.f;
    const float bvv = g.vbias ? g.vbias[cn] : 0.f;
    #pragma unroll
    for (int r = 0; r < 8; ++r) {
      const int m = m0 + half * 8 + r;       // ISA C layout
      float mu  = accM[nt][r] + bmu;
      float var = accV[nt][r] + bvv;
      var = fmaxf(var, TOL);                 // linear/quadratic clamp
      mu  *= g.scale; var *= s2;             // attention 1/sqrt(dk) scaling
      if (g.relu) { if (!(mu > 0.f)) { mu = 0.f; var = TOL; } }
      if (g.pos) mu += g.pos[(long)(m % g.Tmod) * g.N + cn];
      const long off = (long)m * g.ldc + cn;
      if (Rm) { mu += Rm[off]; var = fmaxf(var + Rv[off], TOL); }
      Cm[off] = mu;
      Cv[off] = var;
    }
  }
}

// ---------------------------------------------------------------------------
// LayerNorm-VDP: one 256-thread block per row of length CK (in-place safe).
// ---------------------------------------------------------------------------
__global__ __launch_bounds__(256) void layernorm_vdp_kernel(float* mu, float* var,
                                                            const float* gamma,
                                                            const float* beta) {
  __shared__ float s1[256], s2[256];
  const long base = (long)blockIdx.x * CK;
  const int  n = threadIdx.x;
  float x  = mu[base + n];
  float vx = var[base + n];
  s1[n] = x; s2[n] = x * x;
  __syncthreads();
  for (int s = 128; s > 0; s >>= 1) {
    if (n < s) { s1[n] += s1[n + s]; s2[n] += s2[n + s]; }
    __syncthreads();
  }
  const float m   = s1[0] * (1.f / CK);
  const float v   = s2[0] * (1.f / CK) - m * m;
  const float inv = rsqrtf(v + LNEPS);
  const float gi  = gamma[n] * inv;
  mu[base + n]  = gi * (x - m) + beta[n];
  var[base + n] = fmaxf(gi * gi * vx, TOL);
}

// ---------------------------------------------------------------------------
// Softmax-VDP: one 256-thread block per row of length CT=512 (2 elems/thread).
// ---------------------------------------------------------------------------
__global__ __launch_bounds__(256) void softmax_vdp_kernel(float* mu, float* var) {
  __shared__ float red[256];
  const long base = (long)blockIdx.x * CT;
  const int  t = threadIdx.x;
  float x0 = mu[base + t], x1 = mu[base + t + 256];
  red[t] = fmaxf(x0, x1);
  __syncthreads();
  for (int s = 128; s > 0; s >>= 1) {
    if (t < s) red[t] = fmaxf(red[t], red[t + s]);
    __syncthreads();
  }
  const float mx = red[0];
  __syncthreads();
  float e0 = __expf(x0 - mx), e1 = __expf(x1 - mx);
  red[t] = e0 + e1;
  __syncthreads();
  for (int s = 128; s > 0; s >>= 1) {
    if (t < s) red[t] += red[t + s];
    __syncthreads();
  }
  const float inv = 1.f / red[0];
  const float p0 = e0 * inv, p1 = e1 * inv;
  const float j0 = p0 * (1.f - p0), j1 = p1 * (1.f - p1);
  mu[base + t]        = p0;
  mu[base + t + 256]  = p1;
  var[base + t]       = fmaxf(j0 * j0 * var[base + t], TOL);
  var[base + t + 256] = fmaxf(j1 * j1 * var[base + t + 256], TOL);
}

// ---------------------------------------------------------------------------
// Pack (W, S) -> bf16 triple (W, W^2+S, S), same layout. Also used for K-side
// of attention (k, vk) -> (k, k^2+vk, vk).
// ---------------------------------------------------------------------------
__global__ __launch_bounds__(256) void pack_vdp_kernel(const float* W, const float* S,
                                                       __bf16* b0, __bf16* b1,
                                                       __bf16* b2, int n) {
  const int i = blockIdx.x * 256 + threadIdx.x;
  if (i >= n) return;
  const float w = W[i], s = S[i];
  b0[i] = (__bf16)w;
  b1[i] = (__bf16)(w * w + s);
  b2[i] = (__bf16)s;
}

// V-side pack with transpose: [B,T,H,dh] -> bf16 [B,H,dh,T] triple.
__global__ __launch_bounds__(256) void pack_vT_kernel(const float* v, const float* vv,
                                                      __bf16* b0, __bf16* b1, __bf16* b2) {
  const long i = (long)blockIdx.x * 256 + threadIdx.x;
  const long total = (long)CB * CH * CDH * CT;
  if (i >= total) return;
  const int t = (int)(i % CT); long r = i / CT;
  const int d = (int)(r % CDH); r /= CDH;
  const int h = (int)(r % CH);
  const int b = (int)(r / CH);
  const long in = ((long)(b * CT + t) * CH + h) * CDH + d;
  const float x = v[in], y = vv[in];
  b0[i] = (__bf16)x;
  b1[i] = (__bf16)(x * x + y);
  b2[i] = (__bf16)y;
}

// Final split_heads reorder: [B,T,H*dh] -> 4 x [B,H,T,dh] concatenated into d_out.
__global__ __launch_bounds__(256) void heads_out_kernel(const float* kmu, const float* kvar,
                                                        const float* vmu, const float* vvar,
                                                        float* out) {
  const long i = (long)blockIdx.x * 256 + threadIdx.x;
  const long total = (long)CB * CH * CT * CDH;
  if (i >= total) return;
  const int d = (int)(i % CDH); long r = i / CDH;
  const int t = (int)(r % CT); r /= CT;
  const int h = (int)(r % CH);
  const int b = (int)(r / CH);
  const long in = ((long)(b * CT + t) * CH + h) * CDH + d;
  out[i]             = kmu[in];
  out[total + i]     = kvar[in];
  out[2 * total + i] = vmu[in];
  out[3 * total + i] = vvar[in];
}

// ---------------------------------------------------------------------------
extern "C" void kernel_launch(void* const* d_in, const int* in_sizes, int n_in,
                              void* d_out, int out_size, void* d_ws, size_t ws_size,
                              hipStream_t stream) {
  if (n_in < 56) return;

  // ---- input index maps (insertion order vs jax-pytree sorted order) ----
  int IX, IPOS, IE0[4], IE1[4];
  int IWQ[3], ISQ[3], IWK[3], ISK[3], IWV[3], ISV[3];
  int IFC[3][4], IN1[3][2], IN2[3][2];
  int IFKW, IFKS, IFVW, IFVS;

  const bool ins = (in_sizes[0] == CB * CT * CDIN);
  if (ins) {
    IX = 0;
    for (int j = 0; j < 4; ++j) { IE0[j] = 1 + j; IE1[j] = 5 + j; }
    for (int i = 0; i < 3; ++i) {
      int b = 9 + 6 * i;
      IWQ[i] = b; ISQ[i] = b + 1; IWK[i] = b + 2; ISK[i] = b + 3; IWV[i] = b + 4; ISV[i] = b + 5;
    }
    for (int i = 0; i < 3; ++i) for (int j = 0; j < 4; ++j) IFC[i][j] = 27 + 4 * i + j;
    for (int i = 0; i < 3; ++i) {
      IN1[i][0] = 39 + 2 * i; IN1[i][1] = 40 + 2 * i;
      IN2[i][0] = 45 + 2 * i; IN2[i][1] = 46 + 2 * i;
    }
    IFKW = 51; IFKS = 52; IFVW = 53; IFVS = 54; IPOS = 55;
  } else {
    for (int i = 0; i < 3; ++i) {
      int b = 6 * i;
      ISK[i] = b; ISQ[i] = b + 1; ISV[i] = b + 2; IWK[i] = b + 3; IWQ[i] = b + 4; IWV[i] = b + 5;
    }
    for (int j = 0; j < 4; ++j) { IE0[j] = 18 + j; IE1[j] = 22 + j; }
    for (int i = 0; i < 3; ++i) for (int j = 0; j < 4; ++j) IFC[i][j] = 26 + 4 * i + j;
    IFKS = 38; IFVS = 39; IFKW = 40; IFVW = 41;
    for (int i = 0; i < 3; ++i) {
      IN1[i][0] = 42 + 2 * i; IN1[i][1] = 43 + 2 * i;
      IN2[i][0] = 48 + 2 * i; IN2[i][1] = 49 + 2 * i;
    }
    IPOS = 54; IX = 55;
  }
  auto F = [&](int i) -> const float* { return (const float*)d_in[i]; };

  // ---- workspace carve ----
  char* wsb = (char*)d_ws;
  size_t off = 0;
  auto alloc = [&](size_t bytes) -> void* {
    void* p = wsb + off;
    off += (bytes + 255) & ~(size_t)255;
    return p;
  };

  float* s_mu  = (float*)alloc((size_t)CB * CH * CT * CT * 4);   // scores mu
  float* s_var = (float*)alloc((size_t)CB * CH * CT * CT * 4);   // scores var
  float* x_mu  = (float*)alloc((size_t)CM * CK * 4);
  float* x_var = (float*)alloc((size_t)CM * CK * 4);
  float* t_mu  = (float*)alloc((size_t)CM * CK * 4);
  float* t_var = (float*)alloc((size_t)CM * CK * 4);
  float* q_mu  = (float*)alloc((size_t)CM * CK * 4);
  float* q_var = (float*)alloc((size_t)CM * CK * 4);
  float* k_mu  = (float*)alloc((size_t)CM * CK * 4);
  float* k_var = (float*)alloc((size_t)CM * CK * 4);
  float* v_mu  = (float*)alloc((size_t)CM * CK * 4);
  float* v_var = (float*)alloc((size_t)CM * CK * 4);
  __bf16* kp0 = (__bf16*)alloc((size_t)CM * CK * 2);
  __bf16* kp1 = (__bf16*)alloc((size_t)CM * CK * 2);
  __bf16* kp2 = (__bf16*)alloc((size_t)CM * CK * 2);
  __bf16* vp0 = (__bf16*)alloc((size_t)CM * CK * 2);
  __bf16* vp1 = (__bf16*)alloc((size_t)CM * CK * 2);
  __bf16* vp2 = (__bf16*)alloc((size_t)CM * CK * 2);

  // emb0 output [4096,512] aliases the (not-yet-used) score buffers
  float* e_mu  = s_mu;
  float* e_var = s_var;

  // ---- weight pack table: 0=emb0, 1=emb1, 2+3i+{q,k,v}, 11+i=fc, 14/15=final ----
  const float* Wsrc[16]; const float* Ssrc[16]; size_t Wn[16];
  Wsrc[0] = F(IE0[0]); Ssrc[0] = F(IE0[1]); Wn[0] = (size_t)CE0 * CDIN;
  Wsrc[1] = F(IE1[0]); Ssrc[1] = F(IE1[1]); Wn[1] = (size_t)CK * CE0;
  for (int i = 0; i < 3; ++i) {
    Wsrc[2 + 3 * i + 0] = F(IWQ[i]); Ssrc[2 + 3 * i + 0] = F(ISQ[i]); Wn[2 + 3 * i + 0] = (size_t)CK * CK;
    Wsrc[2 + 3 * i + 1] = F(IWK[i]); Ssrc[2 + 3 * i + 1] = F(ISK[i]); Wn[2 + 3 * i + 1] = (size_t)CK * CK;
    Wsrc[2 + 3 * i + 2] = F(IWV[i]); Ssrc[2 + 3 * i + 2] = F(ISV[i]); Wn[2 + 3 * i + 2] = (size_t)CK * CK;
    Wsrc[11 + i] = F(IFC[i][0]); Ssrc[11 + i] = F(IFC[i][1]); Wn[11 + i] = (size_t)CK * CK;
  }
  Wsrc[14] = F(IFKW); Ssrc[14] = F(IFKS); Wn[14] = (size_t)CK * CK;
  Wsrc[15] = F(IFVW); Ssrc[15] = F(IFVS); Wn[15] = (size_t)CK * CK;

  __bf16 *wp0[16], *wp1[16], *wp2[16];
  for (int l = 0; l < 16; ++l) {
    wp0[l] = (__bf16*)alloc(Wn[l] * 2);
    wp1[l] = (__bf16*)alloc(Wn[l] * 2);
    wp2[l] = (__bf16*)alloc(Wn[l] * 2);
  }
  for (int l = 0; l < 16; ++l) {
    const int n = (int)Wn[l];
    pack_vdp_kernel<<<(n + 255) / 256, 256, 0, stream>>>(Wsrc[l], Ssrc[l],
                                                         wp0[l], wp1[l], wp2[l], n);
  }

  auto launch_gemm = [&](const GemmVdp& g, int z) {
    if (g.N % 64 == 0) {
      const int tiles = (g.M / 16) * (g.N / 64);
      dim3 grid((unsigned)((tiles + 7) / 8), (unsigned)z, 1);
      gemm_vdp_kernel<4><<<grid, dim3(256), 0, stream>>>(g);
    } else if (g.N % 32 == 0) {
      const int tiles = (g.M / 16) * (g.N / 32);
      dim3 grid((unsigned)((tiles + 7) / 8), (unsigned)z, 1);
      gemm_vdp_kernel<2><<<grid, dim3(256), 0, stream>>>(g);
    } else {
      const int tiles = (g.M / 16) * (g.N / 16);
      dim3 grid((unsigned)((tiles + 7) / 8), (unsigned)z, 1);
      gemm_vdp_kernel<1><<<grid, dim3(256), 0, stream>>>(g);
    }
  };

  // ---- emb0: relu(linear(x, vx=0)) -> e ----
  {
    GemmVdp g{};
    g.Amu = F(IX); g.Avar = nullptr;
    g.B0 = wp0[0]; g.B1 = wp1[0]; g.B2 = wp2[0];
    g.Cmu = e_mu; g.Cvar = e_var;
    g.bias = F(IE0[2]); g.vbias = F(IE0[3]);
    g.M = CM; g.N = CE0; g.Kd = CDIN; g.lda = CDIN; g.ldb = CDIN; g.ldc = CE0;
    g.Hdiv = 1; g.scale = 1.f; g.relu = 1; g.Tmod = 1;
    launch_gemm(g, 1);
  }
  // ---- emb1: linear(e) + pos -> x ----
  {
    GemmVdp g{};
    g.Amu = e_mu; g.Avar = e_var;
    g.B0 = wp0[1]; g.B1 = wp1[1]; g.B2 = wp2[1];
    g.Cmu = x_mu; g.Cvar = x_var;
    g.bias = F(IE1[2]); g.vbias = F(IE1[3]);
    g.pos = F(IPOS); g.Tmod = CT;
    g.M = CM; g.N = CK; g.Kd = CE0; g.lda = CE0; g.ldb = CE0; g.ldc = CK;
    g.Hdiv = 1; g.scale = 1.f;
    launch_gemm(g, 1);
  }

  float* cur_mu = x_mu; float* cur_var = x_var;
  float* oth_mu = t_mu; float* oth_var = t_var;

  for (int i = 0; i < 3; ++i) {
    if (i) layernorm_vdp_kernel<<<CM, 256, 0, stream>>>(cur_mu, cur_var,
                                                        F(IN1[i - 1][0]), F(IN1[i - 1][1]));
    // q, k, v projections (no bias)
    float* pm[3] = {q_mu, k_mu, v_mu};
    float* pv[3] = {q_var, k_var, v_var};
    for (int j = 0; j < 3; ++j) {
      GemmVdp g{};
      g.Amu = cur_mu; g.Avar = cur_var;
      const int l = 2 + 3 * i + j;
      g.B0 = wp0[l]; g.B1 = wp1[l]; g.B2 = wp2[l];
      g.Cmu = pm[j]; g.Cvar = pv[j];
      g.M = CM; g.N = CK; g.Kd = CK; g.lda = CK; g.ldb = CK; g.ldc = CK;
      g.Hdiv = 1; g.scale = 1.f; g.Tmod = 1;
      launch_gemm(g, 1);
    }
    // pack K-side (same layout) and V-side (transposed)
    pack_vdp_kernel<<<(CM * CK + 255) / 256, 256, 0, stream>>>(k_mu, k_var, kp0, kp1, kp2, CM * CK);
    pack_vT_kernel<<<(CM * CK + 255) / 256, 256, 0, stream>>>(v_mu, v_var, vp0, vp1, vp2);

    // scores = quadratic_vdp(q, k^T) / sqrt(dk)    (z = b*H + h)
    {
      GemmVdp g{};
      g.Amu = q_mu; g.Avar = q_var;
      g.B0 = kp0; g.B1 = kp1; g.B2 = kp2;
      g.Cmu = s_mu; g.Cvar = s_var;
      g.M = CT; g.N = CT; g.Kd = CDH; g.lda = CK; g.ldb = CK; g.ldc = CT;
      g.Hdiv = CH;
      g.aO = (long)CT * CK; g.aI = CDH;
      g.bO = (long)CT * CK; g.bI = CDH;
      g.cO = (long)CH * CT * CT; g.cI = (long)CT * CT;
      g.scale = 1.0f / 16.0f;   // 1/sqrt(256)
      g.Tmod = 1;
      launch_gemm(g, CB * CH);
    }
    softmax_vdp_kernel<<<CB * CH * CT, 256, 0, stream>>>(s_mu, s_var);

    // out = quadratic_vdp(p, v) merged heads + residual(x) -> oth
    {
      GemmVdp g{};
      g.Amu = s_mu; g.Avar = s_var;
      g.B0 = vp0; g.B1 = vp1; g.B2 = vp2;
      g.Cmu = oth_mu; g.Cvar = oth_var;
      g.Rmu = cur_mu; g.Rvar = cur_var;
      g.M = CT; g.N = CDH; g.Kd = CT; g.lda = CT; g.ldb = CT; g.ldc = CK;
      g.Hdiv = CH;
      g.aO = (long)CH * CT * CT; g.aI = (long)CT * CT;
      g.bO = (long)CH * CDH * CT; g.bI = (long)CDH * CT;
      g.cO = (long)CT * CK; g.cI = CDH;
      g.scale = 1.f; g.Tmod = 1;
      launch_gemm(g, CB * CH);
    }
    { float* tm = cur_mu; cur_mu = oth_mu; oth_mu = tm;
      float* tv = cur_var; cur_var = oth_var; oth_var = tv; }

    layernorm_vdp_kernel<<<CM, 256, 0, stream>>>(cur_mu, cur_var,
                                                 F(IN2[i][0]), F(IN2[i][1]));
    // fc: residual(x0, relu(linear(x0))) -> oth
    {
      GemmVdp g{};
      g.Amu = cur_mu; g.Avar = cur_var;
      const int l = 11 + i;
      g.B0 = wp0[l]; g.B1 = wp1[l]; g.B2 = wp2[l];
      g.Cmu = oth_mu; g.Cvar = oth_var;
      g.bias = F(IFC[i][2]); g.vbias = F(IFC[i][3]);
      g.Rmu = cur_mu; g.Rvar = cur_var;
      g.relu = 1;
      g.M = CM; g.N = CK; g.Kd = CK; g.lda = CK; g.ldb = CK; g.ldc = CK;
      g.Hdiv = 1; g.scale = 1.f; g.Tmod = 1;
      launch_gemm(g, 1);
    }
    { float* tm = cur_mu; cur_mu = oth_mu; oth_mu = tm;
      float* tv = cur_var; cur_var = oth_var; oth_var = tv; }
  }

  layernorm_vdp_kernel<<<CM, 256, 0, stream>>>(cur_mu, cur_var,
                                               F(IN1[2][0]), F(IN1[2][1]));
  // final k / v projections
  for (int j = 0; j < 2; ++j) {
    GemmVdp g{};
    g.Amu = cur_mu; g.Avar = cur_var;
    const int l = 14 + j;
    g.B0 = wp0[l]; g.B1 = wp1[l]; g.B2 = wp2[l];
    g.Cmu = j ? v_mu : k_mu; g.Cvar = j ? v_var : k_var;
    g.M = CM; g.N = CK; g.Kd = CK; g.lda = CK; g.ldb = CK; g.ldc = CK;
    g.Hdiv = 1; g.scale = 1.f; g.Tmod = 1;
    launch_gemm(g, 1);
  }
  heads_out_kernel<<<(CM * CK + 255) / 256, 256, 0, stream>>>(k_mu, k_var, v_mu, v_var,
                                                              (float*)d_out);
  (void)out_size; (void)ws_size;
}
